// GCN_24446953849254
// MI455X (gfx1250) — compile-verified, matrix operand order checked by reading, not run on hardware
//
#include <hip/hip_runtime.h>
#include <hip/hip_bf16.h>

// ---------------------------------------------------------------------------
// GCN forward for MI455X (gfx1250, wave32).
//  - Dense GEMMs via V_WMMA_F32_16X16X4_F32 (exact f32; GEMM cost is tiny vs.
//    the edge scatter, so no precision downgrade is warranted).
//  - Edge aggregation via global_atomic_add_f32; agg arrays (51-102MB) are
//    L2-resident on the 192MB L2, so scatter traffic stays on-chip.
// ---------------------------------------------------------------------------

typedef float    v2f  __attribute__((ext_vector_type(2)));
typedef float    v8f  __attribute__((ext_vector_type(8)));
typedef _Float16 v16h __attribute__((ext_vector_type(16)));

#if defined(__AMDGCN__) && __has_builtin(__builtin_amdgcn_wmma_f32_16x16x4_f32)
#define HAVE_WMMA_F32 1
#else
#define HAVE_WMMA_F32 0
#endif

// One wave32 computes one 16x16 tile of C = A[M,K] @ B[K,N] (+bias)(+relu).
// ISA A layout (32-bit A 16x4): lanes 0-15 -> M=lane, VGPR{0,1}=K{0,1};
// lanes 16-31 -> same M, K{2,3}. B symmetric with N across lanes.
// C/D: lane<16: vgpr v = C[M=v][N=lane]; lane>=16: C[M=v+8][N=lane-16].
__global__ void __launch_bounds__(256) gcn_wmma_gemm(
    const float* __restrict__ A, const float* __restrict__ B,
    const float* __restrict__ bias, float* __restrict__ C,
    int M, int K, int N, int relu)
{
  const int wave   = (blockIdx.x * blockDim.x + threadIdx.x) >> 5;
  const int lane   = threadIdx.x & 31;
  const int tilesN = N >> 4;
  const int tiles  = (M >> 4) * tilesN;
  if (wave >= tiles) return;                 // wave-uniform: EXEC stays all-1s
  const int tm = wave / tilesN;
  const int tn = wave - tm * tilesN;
  const int row0 = tm << 4, col0 = tn << 4;
  const int half = lane >> 4;                // 0: K+{0,1}, 1: K+{2,3}
  const int l    = lane & 15;
  const int bcol = col0 + l;
  const float* Ap = A + (size_t)(row0 + l) * K;

  v8f acc = {};
#if HAVE_WMMA_F32
  const int kmain = K & ~3;
  for (int k = 0; k < kmain; k += 4) {
    const int ka = k + (half << 1);
    v2f a, b;
    a.x = Ap[ka];
    a.y = Ap[ka + 1];
    b.x = B[(size_t)ka * N + bcol];
    b.y = B[(size_t)(ka + 1) * N + bcol];
    acc = __builtin_amdgcn_wmma_f32_16x16x4_f32(false, a, false, b,
                                                (short)0, acc, false, false);
  }
  if (kmain < K) {                           // K tail (K=78): lane-level guards
    const int ka = kmain + (half << 1);
    v2f a, b;
    a.x = (ka     < K) ? Ap[ka]                       : 0.f;
    a.y = (ka + 1 < K) ? Ap[ka + 1]                   : 0.f;
    b.x = (ka     < K) ? B[(size_t)ka * N + bcol]     : 0.f;
    b.y = (ka + 1 < K) ? B[(size_t)(ka + 1) * N + bcol] : 0.f;
    acc = __builtin_amdgcn_wmma_f32_16x16x4_f32(false, a, false, b,
                                                (short)0, acc, false, false);
  }
#else
  // Fallback: codegen-confirmed f16 WMMA (16-bit A 16x32 layout per ISA).
  for (int k0 = 0; k0 < K; k0 += 32) {
    v16h a, b;
    #pragma unroll
    for (int e = 0; e < 16; ++e) {
      const int kk = k0 + (e & 7) + ((e >> 3) << 4) + (half << 3);
      const float av = (kk < K) ? Ap[kk]                     : 0.f;
      const float bv = (kk < K) ? B[(size_t)kk * N + bcol]   : 0.f;
      a[e] = (_Float16)av;
      b[e] = (_Float16)bv;
    }
    acc = __builtin_amdgcn_wmma_f32_16x16x32_f16(false, a, false, b,
                                                 (short)0, acc, false, false);
  }
#endif

  const float bb = bias ? bias[bcol] : 0.f;
  #pragma unroll
  for (int v = 0; v < 8; ++v) {
    float r = acc[v] + bb;
    if (relu) r = fmaxf(r, 0.f);
    C[(size_t)(row0 + v + (half << 3)) * N + bcol] = r;
  }
}

__global__ void zero_kernel(float4* __restrict__ p, long long n4) {
  long long i = blockIdx.x * (long long)blockDim.x + threadIdx.x;
  if (i < n4) p[i] = make_float4(0.f, 0.f, 0.f, 0.f);
}

__global__ void degree_kernel(const int* __restrict__ dst,
                              float* __restrict__ deg, int nE) {
  int e = blockIdx.x * blockDim.x + threadIdx.x;
  if (e < nE) atomicAdd(&deg[dst[e]], 1.0f);
}

__global__ void dinv_kernel(const float* __restrict__ deg,
                            float* __restrict__ dinv, int n) {
  int i = blockIdx.x * blockDim.x + threadIdx.x;
  if (i < n) dinv[i] = rsqrtf(deg[i] + 1.0f);   // +1 self-loop
}

// One thread per (edge, 4-feature chunk): gather H[src]*norm, atomic-scatter
// into agg[dst]. agg fits in the 192MB L2 -> atomics resolve on-chip.
__global__ void edge_agg_kernel(const float* __restrict__ H,
                                const int* __restrict__ src,
                                const int* __restrict__ dst,
                                const float* __restrict__ dinv,
                                float* __restrict__ agg,
                                int nE, int F, int lgC /* log2(F/4) */) {
  const int c = threadIdx.x & ((1 << lgC) - 1);
  const int e = blockIdx.x * (blockDim.x >> lgC) + (threadIdx.x >> lgC);
  if (e >= nE) return;
  const int s = src[e], d = dst[e];
  const float nrm = dinv[s] * dinv[d];
  const float4 hv = ((const float4*)(H + (size_t)s * F))[c];
  float* o = agg + (size_t)d * F + (c << 2);
  atomicAdd(o + 0, hv.x * nrm);
  atomicAdd(o + 1, hv.y * nrm);
  atomicAdd(o + 2, hv.z * nrm);
  atomicAdd(o + 3, hv.w * nrm);
}

// out = relu(agg + Hlin * dinv^2 + bias)  (in-place safe: 1 thread / element)
__global__ void finalize_kernel(const float* __restrict__ agg,
                                const float* __restrict__ Hlin,
                                const float* __restrict__ dinv,
                                const float* __restrict__ bias,
                                float* __restrict__ out,
                                long long total, int lgF) {
  long long idx = blockIdx.x * (long long)blockDim.x + threadIdx.x;
  if (idx >= total) return;
  const int i = (int)(idx >> lgF);
  const int f = (int)(idx & ((1 << lgF) - 1));
  const float di = dinv[i];
  float v = agg[idx] + Hlin[idx] * di * di + bias[f];
  out[idx] = fmaxf(v, 0.f);
}

// Global max pool: post-ReLU values are >= 0, so IEEE order == integer order
// on the raw bits and a zero-initialized buffer is a correct identity.
__global__ void segmax_kernel(const float* __restrict__ h,
                              const int* __restrict__ batch,
                              float* __restrict__ g, long long total) {
  long long idx = blockIdx.x * (long long)blockDim.x + threadIdx.x;
  if (idx >= total) return;
  const int i = (int)(idx >> 8);          // F = 256
  const int f = (int)(idx & 255);
  atomicMax((int*)&g[(size_t)batch[i] * 256 + f], __float_as_int(h[idx]));
}

// ---------------------------------------------------------------------------

static inline void launch_gemm(const float* A, const float* B, const float* bias,
                               float* C, int M, int K, int N, int relu,
                               hipStream_t stream) {
  const int tiles  = (M >> 4) * (N >> 4);
  const int blocks = (tiles + 7) >> 3;        // 8 waves (256 thr) per block
  gcn_wmma_gemm<<<blocks, 256, 0, stream>>>(A, B, bias, C, M, K, N, relu);
}

static inline void launch_zero(float* p, long long nElem, hipStream_t stream) {
  const long long n4 = nElem >> 2;
  const int blocks = (int)((n4 + 255) >> 8);
  zero_kernel<<<blocks, 256, 0, stream>>>((float4*)p, n4);
}

extern "C" void kernel_launch(void* const* d_in, const int* in_sizes, int n_in,
                              void* d_out, int out_size, void* d_ws, size_t ws_size,
                              hipStream_t stream) {
  const float* x    = (const float*)d_in[0];
  const int*   eidx = (const int*)d_in[1];
  const int*   batch= (const int*)d_in[2];
  const float* W1   = (const float*)d_in[3];
  const float* b1   = (const float*)d_in[4];
  const float* W2   = (const float*)d_in[5];
  const float* b2   = (const float*)d_in[6];
  const float* Wg1  = (const float*)d_in[7];
  const float* bg1  = (const float*)d_in[8];
  const float* Wg2  = (const float*)d_in[9];
  const float* bg2  = (const float*)d_in[10];
  float* out = (float*)d_out;

  const int Nn = in_sizes[0] / 78;     // 100000 nodes (6250 exact 16-row tiles)
  const int E  = in_sizes[1] / 2;      // 1.6M edges
  const int G  = 512;
  const int* src = eidx;
  const int* dst = eidx + E;

  // Workspace layout (~208 MB)
  float* ws   = (float*)d_ws;
  float* deg  = ws;                               // [Nn]
  float* dinv = deg  + Nn;                        // [Nn]
  float* bufA = dinv + Nn;                        // [Nn,256] linear outputs
  float* bufB = bufA + (size_t)Nn * 256;          // [Nn,256] agg / activations
  float* gbuf = bufB + (size_t)Nn * 256;          // [512,256]
  float* g1   = gbuf + (size_t)G * 256;           // [512,1024]

  // --- degrees & normalization ---
  launch_zero(deg, Nn, stream);
  degree_kernel<<<(E + 255) / 256, 256, 0, stream>>>(dst, deg, E);
  dinv_kernel<<<(Nn + 255) / 256, 256, 0, stream>>>(deg, dinv, Nn);

  // --- layer 1: F = 128 ---
  launch_gemm(x, W1, nullptr, bufA, Nn, 78, 128, 0, stream);           // H1lin
  launch_zero(bufB, (long long)Nn * 128, stream);
  edge_agg_kernel<<<(E + 7) / 8, 256, 0, stream>>>(bufA, src, dst, dinv,
                                                   bufB, E, 128, 5);
  finalize_kernel<<<(int)(((long long)Nn * 128 + 255) / 256), 256, 0, stream>>>(
      bufB, bufA, dinv, b1, bufB, (long long)Nn * 128, 7);             // h1

  // --- layer 2: F = 256 ---
  launch_gemm(bufB, W2, nullptr, bufA, Nn, 128, 256, 0, stream);       // H2lin
  launch_zero(bufB, (long long)Nn * 256, stream);
  edge_agg_kernel<<<(E + 3) / 4, 256, 0, stream>>>(bufA, src, dst, dinv,
                                                   bufB, E, 256, 6);
  finalize_kernel<<<(int)(((long long)Nn * 256 + 255) / 256), 256, 0, stream>>>(
      bufB, bufA, dinv, b2, bufB, (long long)Nn * 256, 8);             // h2

  // --- global max pool ---
  launch_zero(gbuf, (long long)G * 256, stream);
  segmax_kernel<<<Nn, 256, 0, stream>>>(bufB, batch, gbuf, (long long)Nn * 256);

  // --- MLP head ---
  launch_gemm(gbuf, Wg1, bg1, g1, G, 256, 1024, 1, stream);            // relu
  launch_gemm(g1, Wg2, bg2, out, G, 1024, 128, 0, stream);             // logits
}
